// WindowAttention_10909216932140
// MI455X (gfx1250) — matrix operand order repaired and checked by reading, not compiled, not run on hardware
//
#include <hip/hip_runtime.h>

// ---------------------------------------------------------------------------
// Swin window attention, fused per-window, bf16 WMMA (CDNA5 / gfx1250, wave32)
// x:[8,256,256,96] f32 -> out same shape.  8192 windows, 1 workgroup each.
// Weights pre-converted to bf16 fragment-major layout (prep kernel); 1/sqrt(24)
// folded into the Q weight rows. LDS = 144,384 B -> 2 workgroups per WGP.
// ---------------------------------------------------------------------------

typedef __attribute__((ext_vector_type(16))) __bf16 v16bf;
typedef __attribute__((ext_vector_type(8)))  float  v8f;

#define THREADS 256
#define NWIN    8192

#define CB  8
#define CH  256
#define CW  256
#define CC  96      // channels
#define WS  8       // window size
#define NH  4       // heads
#define HD  24      // head dim
#define TOK 64      // tokens per window
#define C3  288     // 3*C

#define QKV_FRAGS   (3 * 18)          // [k][nt] fragments, 512 bf16 each
#define PROJ_FRAGS  (3 * 6)
#define FRAG_ELEMS  512               // 32 lanes x 16 bf16
#define PROJ_OFF    (QKV_FRAGS * FRAG_ELEMS)   // 27648

// LDS row strides (elements). Dword strides picked for conflict-free access.
#define XST 98      // x_s   [64][98]  bf16 (dword stride 49)
#define QST 34      // q_s/k_s [4][64][34] bf16 (dword stride 17), K-pad 24..31 zeroed
#define VST 66      // v_s   [4][32][66] bf16 transposed (dword stride 33)
#define OST 98      // o_s   [64][98]  bf16
#define SST 66      // scores[4][64][66] f32; softmax repacks bf16 attn in place
#define AST (2*SST) // attn ushort stride inside the scores buffer

#define SMEM_BYTES ((TOK*XST + 2*NH*TOK*QST + NH*32*VST + TOK*OST)*2 \
                    + NH*TOK*SST*4)   // 144384

// float -> bf16 bits (native RNE cvt on gfx1250 via fptrunc).
static __device__ __forceinline__ unsigned short f2bf(float f) {
  __bf16 h = (__bf16)f;
  return __builtin_bit_cast(unsigned short, h);
}

// A-matrix 16x32 bf16 fragment: lanes 0-15 row M=lane hold
// K = 0..15 contiguous; lanes 16-31 hold K = 16..31 (matches B order).
static __device__ __forceinline__ v16bf
load_a_frag(const unsigned short* base, int stride, int row0, int k0, int lane) {
  int r  = row0 + (lane & 15);
  int kb = k0 + ((lane >> 4) << 4);
  const unsigned* p = (const unsigned*)(base + r * stride + kb);
  union { v16bf v; unsigned u[8]; } f;
#pragma unroll
  for (int j = 0; j < 8; ++j) f.u[j] = p[j];
  return f.v;
}

// B-matrix 32x16 bf16 fragment from LDS: lanes 0-15 col N=lane hold K=0..15
// contiguous, lanes 16-31 hold K=16..31 (same K order as load_a_frag).
static __device__ __forceinline__ v16bf
load_b_frag(const unsigned short* base, int stride, int col0, int k0, int lane) {
  int c  = col0 + (lane & 15);
  int kb = k0 + ((lane >> 4) << 4);
  const unsigned* p = (const unsigned*)(base + c * stride + kb);
  union { v16bf v; unsigned u[8]; } f;
#pragma unroll
  for (int j = 0; j < 8; ++j) f.u[j] = p[j];
  return f.v;
}

// B-fragment from pre-fragmentized global weights: lane-contiguous 32B.
static __device__ __forceinline__ v16bf
load_b_frag_g(const unsigned short* fragbase, int fidx, int lane) {
  return ((const v16bf*)(fragbase + fidx * FRAG_ELEMS))[lane];
}

static __device__ __forceinline__ v8f wmma_bf(v16bf a, v16bf b, v8f c) {
  return __builtin_amdgcn_wmma_f32_16x16x32_bf16(false, a, false, b,
                                                 (short)0, c, false, false);
}

// Convert f32 weights into bf16 WMMA B-fragments; scale Q rows by 1/sqrt(24).
__global__ void prep_weights(const float* __restrict__ qkv_w,
                             const float* __restrict__ proj_w,
                             unsigned short* __restrict__ wout) {
  int i = blockIdx.x * blockDim.x + threadIdx.x;
  if (i >= (QKV_FRAGS + PROJ_FRAGS) * FRAG_ELEMS) return;
  const float* src;
  int ntmod, ii = i;
  bool isqkv = (i < PROJ_OFF);
  if (isqkv) { src = qkv_w;  ntmod = 18; }
  else       { src = proj_w; ntmod = 6; ii = i - PROJ_OFF; }
  int f    = ii >> 9;
  int lane = (ii >> 4) & 31;
  int j    = ii & 15;
  int k    = f / ntmod;
  int nt   = f - k * ntmod;
  int o    = nt * 16 + (lane & 15);
  int c    = k * 32 + ((lane >> 4) << 4) + j;
  float w  = src[o * CC + c];
  if (isqkv && o < CC) w *= 0.20412414523193154f;   // fold scale into Wq
  wout[i] = f2bf(w);
}

__global__ void __launch_bounds__(THREADS, 1)
win_attn_kernel(const float* __restrict__ x,
                const float* __restrict__ qkv_b,
                const float* __restrict__ proj_b,
                const unsigned short* __restrict__ wqkv,   // bf16 frags [3][18][512]
                const unsigned short* __restrict__ wproj,  // bf16 frags [3][6][512]
                float* __restrict__ out) {
  extern __shared__ unsigned char smem_raw[];
  unsigned short* x_s = (unsigned short*)smem_raw;          // [64][XST]
  unsigned short* q_s = x_s + TOK * XST;                    // [4][64][QST]
  unsigned short* k_s = q_s + NH * TOK * QST;               // [4][64][QST]
  unsigned short* v_s = k_s + NH * TOK * QST;               // [4][32][VST] (d-major)
  unsigned short* o_s = v_s + NH * 32 * VST;                // [64][OST]
  float*          s_s = (float*)(o_s + TOK * OST);          // [4][64][SST] f32
  unsigned short* a_s = (unsigned short*)s_s;               // attn bf16, in place

  const int tid  = threadIdx.x;
  const int lane = tid & 31;
  const int wave = tid >> 5;

  const int wid = blockIdx.x;
  const int b   = wid >> 10;
  const int wy  = (wid >> 5) & 31;
  const int wx  = wid & 31;

  // -------- phase 0: stage x window -> bf16 LDS; zero head-dim pads --------
  {
    const float4* xv = (const float4*)x;
    for (int e = tid; e < TOK * (CC / 4); e += THREADS) {
      int t  = e / (CC / 4);
      int c4 = e - t * (CC / 4);
      int h  = wy * WS + (t >> 3);
      int w  = wx * WS + (t & 7);
      long gi = ((long)(b * CH + h) * CW + w) * CC + c4 * 4;
      float4 val = xv[gi >> 2];
      unsigned lo = (unsigned)f2bf(val.x) | ((unsigned)f2bf(val.y) << 16);
      unsigned hi = (unsigned)f2bf(val.z) | ((unsigned)f2bf(val.w) << 16);
      unsigned* dst = (unsigned*)(x_s + t * XST + c4 * 4);
      dst[0] = lo; dst[1] = hi;
    }
    for (int i = tid; i < NH * TOK * (32 - HD); i += THREADS) {  // 2048
      int head = i >> 9;
      int rem  = i & 511;
      int tq   = rem >> 3;
      int dq   = HD + (rem & 7);
      q_s[(head * TOK + tq) * QST + dq] = 0;
      k_s[(head * TOK + tq) * QST + dq] = 0;
      int dv = HD + (rem >> 6);
      int tv = rem & 63;
      v_s[(head * 32 + dv) * VST + tv] = 0;
    }
  }
  __syncthreads();

  // -------- phase 1: qkv = x @ Wqkv^T + b  (scatter into q/k/v layouts) ----
  const float scale = 0.20412414523193154f;  // 1/sqrt(24), already in Wq
  for (int nt = wave; nt < C3 / 16; nt += 8) {
    v8f acc[4] = { {}, {}, {}, {} };
#pragma unroll
    for (int k = 0; k < 3; ++k) {
      v16bf bf = load_b_frag_g(wqkv, k * 18 + nt, lane);
#pragma unroll
      for (int m = 0; m < 4; ++m) {
        v16bf af = load_a_frag(x_s, XST, m * 16, k * 32, lane);
        acc[m] = wmma_bf(af, bf, acc[m]);
      }
    }
    int   baseo = nt * 16;
    int   which = baseo / CC;               // uniform: 96 is a multiple of 16
    float bias  = qkv_b[baseo + (lane & 15)];
    if (which == 0) bias *= scale;          // match scaled Wq
    int   lrem  = baseo - which * CC + (lane & 15);
    int   head  = (lrem * 43) >> 10;        // exact /24 for lrem < 96
    int   d     = lrem - head * HD;
    int   rbase = 8 * (lane >> 4);
    if (which < 2) {                        // q and k share layout
      unsigned short* bp = ((which == 0) ? q_s : k_s) + head * TOK * QST + d;
#pragma unroll
      for (int m = 0; m < 4; ++m)
#pragma unroll
        for (int r = 0; r < 8; ++r)
          bp[(m * 16 + rbase + r) * QST] = f2bf(acc[m][r] + bias);
    } else {
      unsigned short* vp = v_s + (head * 32 + d) * VST;
#pragma unroll
      for (int m = 0; m < 4; ++m)
#pragma unroll
        for (int r = 0; r < 8; ++r)
          vp[m * 16 + rbase + r] = f2bf(acc[m][r] + bias);
    }
  }
  __syncthreads();

  // -------- phase 2: scores = q @ k^T  (K = 32 padded head dim) ------------
  {
    int head = wave >> 1;
    int half = wave & 1;
    const unsigned short* qh = q_s + head * TOK * QST;
    const unsigned short* kh = k_s + head * TOK * QST;
    float* sh = s_s + head * TOK * SST;
    int rbase = 8 * (lane >> 4);
    int col16 = lane & 15;
#pragma unroll
    for (int mi = 0; mi < 2; ++mi) {
      int m = half * 2 + mi;
      v16bf af = load_a_frag(qh, QST, m * 16, 0, lane);
#pragma unroll
      for (int n = 0; n < 4; ++n) {
        v16bf bf = load_b_frag(kh, QST, n * 16, 0, lane);
        v8f acc = {};
        acc = wmma_bf(af, bf, acc);
        int col = n * 16 + col16;
#pragma unroll
        for (int r = 0; r < 8; ++r)
          sh[(m * 16 + rbase + r) * SST + col] = acc[r];
      }
    }
  }
  __syncthreads();

  // -------- softmax: one thread per (head,row); pack bf16 attn in place ----
  {
    int head = tid >> 6;
    int row  = tid & 63;
    float* sr = s_s + (head * TOK + row) * SST;
    const float2* sr2 = (const float2*)sr;
    float mx = -3.4e38f;
    for (int j2 = 0; j2 < TOK / 2; ++j2) {
      float2 e = sr2[j2];
      mx = fmaxf(mx, fmaxf(e.x, e.y));
    }
    float sum = 0.f;
    for (int j2 = 0; j2 < TOK / 2; ++j2) {
      float2 e = sr2[j2];
      float e0 = __expf(e.x - mx), e1 = __expf(e.y - mx);
      sum += e0 + e1;
      ((float2*)sr)[j2] = make_float2(e0, e1);
    }
    float inv = 1.0f / sum;
    unsigned* arw = (unsigned*)sr;          // overwrite consumed f32 with bf16
    for (int j2 = 0; j2 < TOK / 2; ++j2) {
      float2 e = sr2[j2];                   // read dwords 2*j2, 2*j2+1
      unsigned pk = (unsigned)f2bf(e.x * inv) |
                    ((unsigned)f2bf(e.y * inv) << 16);
      arw[j2] = pk;                         // write dword j2 <= 2*j2 (safe)
    }
  }
  __syncthreads();

  // -------- phase 3: outh = attn @ v  (K = 64, two WMMA K-steps) ------------
  {
    int head = wave >> 1;
    int half = wave & 1;
    const unsigned short* ah = a_s + head * TOK * AST;
    const unsigned short* vh = v_s + head * 32 * VST;
#pragma unroll
    for (int mi = 0; mi < 2; ++mi) {
      int m = half * 2 + mi;
      v8f acc[2] = { {}, {} };
#pragma unroll
      for (int k = 0; k < 2; ++k) {
        v16bf af = load_a_frag(ah, AST, m * 16, k * 32, lane);
#pragma unroll
        for (int n = 0; n < 2; ++n) {
          v16bf bf = load_b_frag(vh, VST, n * 16, k * 32, lane);
          acc[n] = wmma_bf(af, bf, acc[n]);
        }
      }
      int rbase = 8 * (lane >> 4);
#pragma unroll
      for (int n = 0; n < 2; ++n) {
        int d = n * 16 + (lane & 15);
        if (d < HD) {
          int c = head * HD + d;
#pragma unroll
          for (int r = 0; r < 8; ++r)
            o_s[(m * 16 + rbase + r) * OST + c] = f2bf(acc[n][r]);
        }
      }
    }
  }
  __syncthreads();

  // -------- phase 4: out = o @ Wproj^T + b, window-reverse store ------------
  for (int job = wave; job < 24; job += 8) {
    int nt = job >> 2;
    int m  = job & 3;
    v8f acc = {};
#pragma unroll
    for (int k = 0; k < 3; ++k) {
      v16bf af = load_a_frag(o_s, OST, m * 16, k * 32, lane);
      v16bf bf = load_b_frag_g(wproj, k * 6 + nt, lane);
      acc = wmma_bf(af, bf, acc);
    }
    int   c     = nt * 16 + (lane & 15);
    float bias  = proj_b[c];
    int   rbase = 8 * (lane >> 4);
#pragma unroll
    for (int r = 0; r < 8; ++r) {
      int token = m * 16 + rbase + r;
      int h = wy * WS + (token >> 3);
      int w = wx * WS + (token & 7);
      out[((long)(b * CH + h) * CW + w) * CC + c] = acc[r] + bias;
    }
  }
}

extern "C" void kernel_launch(void* const* d_in, const int* in_sizes, int n_in,
                              void* d_out, int out_size, void* d_ws, size_t ws_size,
                              hipStream_t stream) {
  const float* x      = (const float*)d_in[0];
  const float* qkv_b  = (const float*)d_in[2];
  const float* proj_b = (const float*)d_in[4];
  unsigned short* wbf = (unsigned short*)d_ws;   // bf16 fragment weights

  int total_w = (QKV_FRAGS + PROJ_FRAGS) * FRAG_ELEMS;   // 36864
  prep_weights<<<(total_w + THREADS - 1) / THREADS, THREADS, 0, stream>>>(
      (const float*)d_in[1], (const float*)d_in[3], wbf);

  (void)hipFuncSetAttribute((const void*)win_attn_kernel,
                            hipFuncAttributeMaxDynamicSharedMemorySize,
                            (int)SMEM_BYTES);
  win_attn_kernel<<<NWIN, THREADS, SMEM_BYTES, stream>>>(
      x, qkv_b, proj_b, wbf, wbf + PROJ_OFF, (float*)d_out);
}
